// GrowableRotationEngine_15229954031690
// MI455X (gfx1250) — compile-verified, matrix operand order checked by reading, not run on hardware
//
#include <hip/hip_runtime.h>

// ---------------- CDNA5 / gfx1250 types ----------------
typedef __attribute__((ext_vector_type(16))) __bf16 v16bf;
typedef __attribute__((ext_vector_type(8)))  __bf16 v8bf;
typedef __attribute__((ext_vector_type(4)))  __bf16 v4bf;
typedef __attribute__((ext_vector_type(8)))  float  v8f;
typedef __attribute__((ext_vector_type(4)))  int    v4i;

// ---------------- problem constants ----------------
constexpr int   Bb  = 2;
constexpr int   Ss  = 2048;
constexpr int   Dd  = 1024;
constexpr int   Vv  = 32000;
constexpr int   Cc  = 8;
constexpr int   Pp  = 256;
constexpr float kEps = 1.1920929e-07f;

// GEMM tiling: block = 256 threads = 8 wave32s.
// Block tile 64x64; each wave owns a 16x32 C tile (2 WMMAs sharing one A frag).
constexpr int BM  = 64;
constexpr int BN  = 64;
constexpr int BK  = 32;
constexpr int LDA = 40;   // padded LDS row stride (bf16) -> conflict-free b128 reads
constexpr int LDB = 40;
constexpr int NT  = Dd / BK;   // 32 k-steps

// ---------------- gfx1250 async-to-LDS path (probe-gated) ----------------
#ifndef __has_builtin
#define __has_builtin(x) 0
#endif
#if __has_builtin(__builtin_amdgcn_global_load_async_to_lds_b128)
#define GRE_ASYNC 1
#else
#define GRE_ASYNC 0
#endif

#if GRE_ASYNC
typedef __attribute__((address_space(1))) void  gv_t;
typedef __attribute__((address_space(3))) void  lv_t;
typedef __attribute__((address_space(1))) v4i   gv4i_t;
typedef __attribute__((address_space(3))) v4i   lv4i_t;
static __device__ __forceinline__ void gre_async_b128(const void* g, void* l) {
  // GLOBAL_LOAD_ASYNC_TO_LDS_B128: per-lane global -> LDS, tracked by ASYNCcnt.
  // Param types per hipcc diagnostic: (AS1 v4i*, AS3 v4i*, imm, imm).
  gv_t* g1 = (gv_t*)g;   // generic -> AS1 (addrspacecast)
  lv_t* l3 = (lv_t*)l;   // generic -> AS3 (addrspacecast)
  __builtin_amdgcn_global_load_async_to_lds_b128((gv4i_t*)g1, (lv4i_t*)l3, 0, 0);
}
#endif

static __device__ __forceinline__ void gre_wait_async0() {
#if __has_builtin(__builtin_amdgcn_s_wait_asynccnt)
  __builtin_amdgcn_s_wait_asynccnt(0);
#elif GRE_ASYNC
  asm volatile("s_wait_asynccnt 0" ::: "memory");
#endif
}

// ---------------- kernel 1: lm_head f32 -> bf16 ----------------
__global__ __launch_bounds__(256)
void gre_conv_w(const float* __restrict__ W, __bf16* __restrict__ Wb) {
  size_t idx = ((size_t)blockIdx.x * 256 + threadIdx.x) * 4;
  float4 f = *(const float4*)(W + idx);
  v4bf o;
  o[0] = (__bf16)f.x; o[1] = (__bf16)f.y; o[2] = (__bf16)f.z; o[3] = (__bf16)f.w;
  *(v4bf*)(Wb + idx) = o;
}

// ---------------- kernel 2: embedding gather ----------------
__global__ __launch_bounds__(256)
void gre_embed(const int* __restrict__ tokens, const float* __restrict__ E,
               float* __restrict__ X) {
  int t   = blockIdx.x;
  int tok = tokens[t];
  int d   = threadIdx.x * 4;
  *(float4*)(X + (size_t)t * Dd + d) = *(const float4*)(E + (size_t)tok * Dd + d);
}

// ---------------- kernel 3: one cycle (mix + rotate + silu + rmsnorm residual) ----
__global__ __launch_bounds__(256)
void gre_cycle(const float* __restrict__ Xin, float* __restrict__ Xout,
               const float* __restrict__ angles, const float* __restrict__ scales,
               const float* __restrict__ shifts, const float* __restrict__ nweights,
               const float* __restrict__ mixw, const int* __restrict__ pi,
               const int* __restrict__ pj, int c) {
  __shared__ float sm[Dd];    // mixed x (pre-rotation residual branch)
  __shared__ float sr[Dd];    // rotated x
  __shared__ float red[256];

  const int t   = blockIdx.x;
  const int s   = t & (Ss - 1);
  const int tid = threadIdx.x;

  const float mw0 = mixw[0], mw1 = mixw[1], mw2 = mixw[2];
  const float* xc = Xin + (size_t)t * Dd;

#pragma unroll
  for (int i = 0; i < 4; ++i) {
    int d = tid + i * 256;
    float v = mw1 * xc[d];
    if (s > 0)      v += mw0 * xc[d - Dd];
    if (s < Ss - 1) v += mw2 * xc[d + Dd];
    sm[d] = v;
    sr[d] = v;
  }
  __syncthreads();

  // Givens rotation on disjoint plane pairs: one plane per thread (P == 256).
  {
    int   ip = pi[c * Pp + tid];
    int   jp = pj[c * Pp + tid];
    float sa, ca;
    sincosf(angles[c * Pp + tid], &sa, &ca);
    float xi = sm[ip], xj = sm[jp];
    sr[ip] = xi * ca - xj * sa;
    sr[jp] = xi * sa + xj * ca;
  }
  __syncthreads();

  const float* sc = scales   + (size_t)c * Dd;
  const float* sh = shifts   + (size_t)c * Dd;
  const float* nw = nweights + (size_t)c * Dd;

  float part = 0.f;
  float vdiff[4];
#pragma unroll
  for (int i = 0; i < 4; ++i) {
    int d = tid + i * 256;
    float a   = sr[d] * sc[d] + sh[d];
    float act = a / (1.f + __expf(-a));     // silu
    float v   = act - sm[d];                // residual is vs. MIXED x (pre-rotation)
    vdiff[i]  = v;
    part     += v * v;
  }
  red[tid] = part;
  __syncthreads();
  for (int off = 128; off > 0; off >>= 1) {
    if (tid < off) red[tid] += red[tid + off];
    __syncthreads();
  }
  float rs = rsqrtf(red[0] * (1.f / (float)Dd) + kEps);

  float* xo = Xout + (size_t)t * Dd;
#pragma unroll
  for (int i = 0; i < 4; ++i) {
    int d = tid + i * 256;
    xo[d] = sm[d] + vdiff[i] * rs * nw[d];
  }
}

// ---------------- kernel 4: final RMSNorm -> bf16 activations ----------------
__global__ __launch_bounds__(256)
void gre_final_norm(const float* __restrict__ Xin, const float* __restrict__ w,
                    __bf16* __restrict__ Aout) {
  __shared__ float red[256];
  const int t   = blockIdx.x;
  const int tid = threadIdx.x;
  const float* x = Xin + (size_t)t * Dd;

  float v[4];
  float part = 0.f;
#pragma unroll
  for (int i = 0; i < 4; ++i) {
    int d = tid + i * 256;
    v[i]  = x[d];
    part += v[i] * v[i];
  }
  red[tid] = part;
  __syncthreads();
  for (int off = 128; off > 0; off >>= 1) {
    if (tid < off) red[tid] += red[tid + off];
    __syncthreads();
  }
  float rs = rsqrtf(red[0] * (1.f / (float)Dd) + kEps);

#pragma unroll
  for (int i = 0; i < 4; ++i) {
    int d = tid + i * 256;
    Aout[(size_t)t * Dd + d] = (__bf16)(v[i] * rs * w[d]);
  }
}

// ---------------- kernel 5: logits GEMM, bf16 WMMA f32-acc ----------------
// C[M=4096, N=32000] = A[M,K] * W[N,K]^T
__global__ __launch_bounds__(256)
void gre_gemm_bf16(const __bf16* __restrict__ A, const __bf16* __restrict__ Wb,
                   float* __restrict__ out) {
  __shared__ __bf16 As[2][BM * LDA];
  __shared__ __bf16 Bs[2][BN * LDB];

  const int tid = threadIdx.x;
  const int m0  = blockIdx.y * BM;
  const int n0  = blockIdx.x * BN;

  // cooperative loaders: both tiles are 64 rows x 32 k -> one 16B chunk/thread
  const int lrow = tid >> 2;
  const int lchk = (tid & 3) * 8;
  const size_t abase = (size_t)(m0 + lrow) * Dd + lchk;
  const size_t bbase = (size_t)(n0 + lrow) * Dd + lchk;
  const int    loff  = lrow * LDA + lchk;

#if GRE_ASYNC
  gre_async_b128(A  + abase, &As[0][loff]);
  gre_async_b128(Wb + bbase, &Bs[0][loff]);
  gre_wait_async0();
#else
  {
    uint4 a0 = *(const uint4*)(A  + abase);
    uint4 b0 = *(const uint4*)(Wb + bbase);
    *(uint4*)&As[0][loff] = a0;
    *(uint4*)&Bs[0][loff] = b0;
  }
#endif
  __syncthreads();

  const int lane  = tid & 31;
  const int wave  = tid >> 5;
  const int mt    = (wave & 3) * 16;           // 4 row tiles
  const int ng    = (wave >> 2) * 32;          // 2 col groups of 32
  const int mrow  = mt + (lane & 15);
  const int n0row = ng + (lane & 15);
  const int n1row = n0row + 16;
  const int kbase = (lane >> 4) * 8;           // 16-bit A/B layout: hi lanes hold K+8

  v8f acc0 = {}, acc1 = {};
#if !GRE_ASYNC
  uint4 areg, breg;
#endif

  for (int kt = 0; kt < NT; ++kt) {
    const int cur = kt & 1;
    const int nxt = cur ^ 1;

    if (kt + 1 < NT) {
      const size_t koff = (size_t)(kt + 1) * BK;
      if (kt + 2 < NT) {
        __builtin_prefetch(A  + abase + koff + BK, 0, 3);   // global_prefetch_b8
        __builtin_prefetch(Wb + bbase + koff + BK, 0, 3);
      }
#if GRE_ASYNC
      gre_async_b128(A  + abase + koff, &As[nxt][loff]);    // ASYNCcnt-tracked
      gre_async_b128(Wb + bbase + koff, &Bs[nxt][loff]);
#else
      areg = *(const uint4*)(A  + abase + koff);
      breg = *(const uint4*)(Wb + bbase + koff);
#endif
    }

    // fragments: elems 0..7 <- K=kbase..kbase+7, elems 8..15 <- K=16+kbase..
    v8bf alo  = *(const v8bf*)&As[cur][mrow  * LDA + kbase];
    v8bf ahi  = *(const v8bf*)&As[cur][mrow  * LDA + kbase + 16];
    v8bf b0lo = *(const v8bf*)&Bs[cur][n0row * LDB + kbase];
    v8bf b0hi = *(const v8bf*)&Bs[cur][n0row * LDB + kbase + 16];
    v8bf b1lo = *(const v8bf*)&Bs[cur][n1row * LDB + kbase];
    v8bf b1hi = *(const v8bf*)&Bs[cur][n1row * LDB + kbase + 16];
    v16bf af, bf0, bf1;
#pragma unroll
    for (int i = 0; i < 8; ++i) {
      af[i]  = alo[i];  af[8 + i]  = ahi[i];
      bf0[i] = b0lo[i]; bf0[8 + i] = b0hi[i];
      bf1[i] = b1lo[i]; bf1[8 + i] = b1hi[i];
    }

    acc0 = __builtin_amdgcn_wmma_f32_16x16x32_bf16(
        false, af, false, bf0, (short)0, acc0, false, false);
    acc1 = __builtin_amdgcn_wmma_f32_16x16x32_bf16(
        false, af, false, bf1, (short)0, acc1, false, false);

    if (kt + 1 < NT) {
#if GRE_ASYNC
      gre_wait_async0();        // this wave's async tile writes complete
#else
      *(uint4*)&As[nxt][loff] = areg;
      *(uint4*)&Bs[nxt][loff] = breg;
#endif
      __syncthreads();          // all waves' tile writes visible
    }
  }

  // epilogue: VGPR r -> M=r (lanes 0-15) / M=8+r (lanes 16-31), N=lane&15
  const int rb   = (lane >> 4) * 8;
  const int col0 = n0 + ng + (lane & 15);
#pragma unroll
  for (int r = 0; r < 8; ++r) {
    int row = m0 + mt + rb + r;
    float* o = out + (size_t)row * Vv + col0;
    o[0]  = acc0[r];
    o[16] = acc1[r];
  }
}

// ---------------- host side ----------------
extern "C" void kernel_launch(void* const* d_in, const int* in_sizes, int n_in,
                              void* d_out, int out_size, void* d_ws, size_t ws_size,
                              hipStream_t stream) {
  (void)in_sizes; (void)n_in; (void)out_size; (void)ws_size;

  const int*   tokens   = (const int*)  d_in[0];
  const float* embed_w  = (const float*)d_in[1];
  const float* angles   = (const float*)d_in[2];
  const float* scales   = (const float*)d_in[3];
  const float* shifts   = (const float*)d_in[4];
  const float* nweights = (const float*)d_in[5];
  const float* mixw     = (const float*)d_in[6];
  const float* onw      = (const float*)d_in[7];
  const float* lmh      = (const float*)d_in[8];
  const int*   pi       = (const int*)  d_in[9];
  const int*   pj       = (const int*)  d_in[10];
  float*       out      = (float*)d_out;

  char* ws = (char*)d_ws;
  float*  Xa  = (float*)(ws);                                   // 16 MiB
  float*  Xb  = (float*)(ws + (size_t)16 * 1024 * 1024);        // 16 MiB
  __bf16* Abf = (__bf16*)(ws + (size_t)32 * 1024 * 1024);       // 8 MiB
  __bf16* Wbf = (__bf16*)(ws + (size_t)40 * 1024 * 1024);       // 62.5 MiB

  const int M = Bb * Ss;  // 4096 tokens

  gre_conv_w<<<Vv, 256, 0, stream>>>(lmh, Wbf);
  gre_embed<<<M, 256, 0, stream>>>(tokens, embed_w, Xa);

  for (int c = 0; c < Cc; ++c) {
    const float* xin = (c & 1) ? Xb : Xa;
    float*       xo  = (c & 1) ? Xa : Xb;
    gre_cycle<<<M, 256, 0, stream>>>(xin, xo, angles, scales, shifts, nweights,
                                     mixw, pi, pj, c);
  }
  // C=8 cycles -> result lands back in Xa
  gre_final_norm<<<M, 256, 0, stream>>>(Xa, onw, Abf);

  dim3 grid(Vv / BN, M / BM);   // 500 x 64 blocks
  gre_gemm_bf16<<<grid, 256, 0, stream>>>(Abf, Wbf, out);
}
